// StyleAttentionLayer_69784628626090
// MI455X (gfx1250) — matrix operand order repaired and checked by reading, not compiled, not run on hardware
//
#include <hip/hip_runtime.h>

typedef __bf16 bf16;
typedef bf16  v16bf __attribute__((ext_vector_type(16)));
typedef float v8f   __attribute__((ext_vector_type(8)));
typedef int   v4i   __attribute__((vector_size(16)));

#define ATTN_SCALE 0.04419417382415922f  // 512^-0.5

// ---- gfx1250 async global->LDS copy (ASYNCcnt-tracked), with fallback ----
#if defined(__has_builtin)
#if __has_builtin(__builtin_amdgcn_global_load_async_to_lds_b128) && \
    __has_builtin(__builtin_amdgcn_s_wait_asynccnt)
#define USE_ASYNC_LDS 1
#endif
#endif
#ifndef USE_ASYNC_LDS
#define USE_ASYNC_LDS 0
#endif

__device__ __forceinline__ void wait_async_done() {
#if USE_ASYNC_LDS
  __builtin_amdgcn_s_wait_asynccnt(0);
#endif
}

// Copy one 64KB weight slice (64 rows x 512 bf16) global -> LDS, all 256
// threads cooperating; 16B per thread per step, 4KB contiguous per step.
__device__ __forceinline__ void stage_slice(const bf16* __restrict__ g, bf16* l,
                                            int tid) {
#if USE_ASYNC_LDS
  typedef __attribute__((address_space(1))) v4i* gv4p;
  typedef __attribute__((address_space(3))) v4i* lv4p;
#endif
#pragma unroll
  for (int c = 0; c < 16; ++c) {
    int o = c * 4096 + tid * 16;  // bytes
#if USE_ASYNC_LDS
    __builtin_amdgcn_global_load_async_to_lds_b128(
        (gv4p)((const char*)g + o), (lv4p)((char*)l + o), 0, 0);
#else
    *(uint4*)((char*)l + o) = *(const uint4*)((const char*)g + o);
#endif
  }
}

__device__ __forceinline__ v8f wmma_bf16(v16bf a, v16bf b, v8f c) {
  // D = A(16x32) * B(32x16) + C, f32 accumulate
  return __builtin_amdgcn_wmma_f32_16x16x32_bf16(false, a, false, b, (short)0, c,
                                                 false, false);
}

__device__ __forceinline__ v8f zero8() {
  v8f z = {0.f, 0.f, 0.f, 0.f, 0.f, 0.f, 0.f, 0.f};
  return z;
}

// Build a 16-bit A-matrix fragment (16x32, this lane's share) from an f32 row.
// Lanes 0-15: row M=lane, K in {0..15}; lanes 16-31: same rows, K in {16..31}.
// Caller passes p = row_base + kb (kb = 16*(lane>>4)).
__device__ __forceinline__ v16bf make_afrag_f32(const float* p) {
  float4 c0 = *(const float4*)(p + 0);
  float4 c1 = *(const float4*)(p + 4);
  float4 c2 = *(const float4*)(p + 8);
  float4 c3 = *(const float4*)(p + 12);
  v16bf a;
  a[0]  = (bf16)c0.x; a[1]  = (bf16)c0.y; a[2]  = (bf16)c0.z; a[3]  = (bf16)c0.w;
  a[4]  = (bf16)c1.x; a[5]  = (bf16)c1.y; a[6]  = (bf16)c1.z; a[7]  = (bf16)c1.w;
  a[8]  = (bf16)c2.x; a[9]  = (bf16)c2.y; a[10] = (bf16)c2.z; a[11] = (bf16)c2.w;
  a[12] = (bf16)c3.x; a[13] = (bf16)c3.y; a[14] = (bf16)c3.z; a[15] = (bf16)c3.w;
  return a;
}

// Store 4 C-layout tiles (16x64 f32) into per-wave LDS row-major buffer (ld=64).
__device__ __forceinline__ void stage4(float* buf, const v8f* t, int lrow, int lhalf) {
#pragma unroll
  for (int nt = 0; nt < 4; ++nt)
#pragma unroll
    for (int r = 0; r < 8; ++r)
      buf[(r + 8 * lhalf) * 64 + nt * 16 + lrow] = t[nt][r];
}

// ---------------- prep kernels (tiny cost) ----------------

__global__ void prep_weights(const float* __restrict__ Wq, const float* __restrict__ Wo,
                             bf16* __restrict__ WqT, bf16* __restrict__ WoT) {
  int tid = blockIdx.x * blockDim.x + threadIdx.x;  // 0..262143 = n*512 + k
  int n = tid >> 9, k = tid & 511;
  WqT[tid] = (bf16)Wq[k * 512 + n];
  WoT[tid] = (bf16)Wo[k * 512 + n];
}

// key_const (H,1,64,50) -> kT[h][s][d] = key*SCALE, s padded 50->64 with 0.
__global__ void prep_keys(const float* __restrict__ keyc, bf16* __restrict__ kT) {
  int tid = blockIdx.x * blockDim.x + threadIdx.x;  // 0..32767 = h*4096 + s*64 + d
  int h = tid >> 12, s = (tid >> 6) & 63, d = tid & 63;
  float v = 0.f;
  if (s < 50) v = keyc[(h * 64 + d) * 50 + s] * ATTN_SCALE;
  kT[tid] = (bf16)v;
}

// v = values@Wv + bv, stored vb[h][b][d][s] (s padded 50->64 with 0), bf16.
__global__ void prep_values(const float* __restrict__ values, const float* __restrict__ Wv,
                            const float* __restrict__ bv, bf16* __restrict__ vb) {
  int tid = blockIdx.x * blockDim.x + threadIdx.x;  // 0..524287
  int s = tid & 63;
  int d = (tid >> 6) & 63;
  int b = (tid >> 12) & 15;
  int h = tid >> 16;
  float acc = 0.f;
  if (s < 50) {
    int n = h * 64 + d;
    acc = bv[n];
    const float* vr = values + (b * 50 + s) * 128;
#pragma unroll 8
    for (int k = 0; k < 128; ++k) acc = fmaf(vr[k], Wv[k * 512 + n], acc);
  }
  vb[tid] = (bf16)acc;
}

// ---------------- fused main kernel ----------------
// grid = 512 blocks x 256 threads (8 waves), 128 rows per block, 16 per wave.
// 16 pipeline stages per block: 8 heads (Wq slices) + 8 output groups (Wo
// slices). Each 64KB weight slice is staged once per block into double-buffered
// dynamic LDS via async global->LDS copies, prefetching slice i+1 during
// compute of slice i.

extern __shared__ char dynlds[];  // 2 x 64KB weight slice double buffer

__global__ __launch_bounds__(256) void style_attn_fused(
    const float* __restrict__ x, const float* __restrict__ mask,
    const float* __restrict__ bq, const float* __restrict__ bo,
    const bf16* __restrict__ WqT, const bf16* __restrict__ WoT,
    const bf16* __restrict__ kT, const bf16* __restrict__ vbuf,
    float* __restrict__ out) {
  __shared__ __align__(16) float sbuf[8 * 16 * 64];  // 4KB per wave, private

  const int tid = threadIdx.x;
  const int lane = tid & 31;
  const int wid = tid >> 5;
  const int lrow = lane & 15;
  const int lhalf = lane >> 4;
  const int kb = lhalf * 16;
  const int rowbase = (blockIdx.x * 8 + wid) * 16;  // 16 rows per wave
  const int batch = rowbase >> 12;                  // row / 4096
  float* buf = sbuf + wid * 1024;

  bf16* wb0 = (bf16*)dynlds;
  bf16* wb1 = (bf16*)(dynlds + 65536);

  // kick off stage 0 (Wq head 0) before doing local prework
  stage_slice(WqT, wb0, tid);

  // ---- preload x A-fragments: 16 rows x 512 K as bf16 (128 VGPRs) ----
  const float* xrow = x + (size_t)(rowbase + lrow) * 512;
  v16bf ax[16];
#pragma unroll
  for (int kc = 0; kc < 16; ++kc) ax[kc] = make_afrag_f32(xrow + kc * 32 + kb);

  float mrow[8];
#pragma unroll
  for (int r = 0; r < 8; ++r) mrow[r] = mask[rowbase + r + 8 * lhalf];

  v16bf ctxa[16];  // ctx(16x512) A-fragments, filled 2 per head

#pragma unroll
  for (int h = 0; h < 8; ++h) {
    bf16* cur = (h & 1) ? wb1 : wb0;
    bf16* nxt = (h & 1) ? wb0 : wb1;
    wait_async_done();   // our portion of slice h landed
    __syncthreads();     // slice h visible; buffer 'nxt' no longer read
    // prefetch next slice: Wq head h+1, or Wo group 0 after the last head
    stage_slice((h < 7) ? (WqT + (size_t)(h + 1) * 32768) : WoT, nxt, tid);

    // ---- q_h = x @ Wq[:, 64h:64h+64] + bq  (B from LDS slice) ----
    v8f qt[4];
#pragma unroll
    for (int nt = 0; nt < 4; ++nt) qt[nt] = zero8();
    {
      const bf16* wbp = cur + (size_t)lrow * 512 + lhalf * 16;
#pragma unroll
      for (int kc = 0; kc < 16; ++kc) {
        v16bf b0 = *(const v16bf*)(wbp + (0 * 16) * 512 + kc * 32);
        v16bf b1 = *(const v16bf*)(wbp + (1 * 16) * 512 + kc * 32);
        v16bf b2 = *(const v16bf*)(wbp + (2 * 16) * 512 + kc * 32);
        v16bf b3 = *(const v16bf*)(wbp + (3 * 16) * 512 + kc * 32);
        qt[0] = wmma_bf16(ax[kc], b0, qt[0]);
        qt[1] = wmma_bf16(ax[kc], b1, qt[1]);
        qt[2] = wmma_bf16(ax[kc], b2, qt[2]);
        qt[3] = wmma_bf16(ax[kc], b3, qt[3]);
      }
    }
#pragma unroll
    for (int nt = 0; nt < 4; ++nt) {
      float bqv = bq[h * 64 + nt * 16 + lrow];
#pragma unroll
      for (int r = 0; r < 8; ++r) qt[nt][r] += bqv;
    }

    // C-layout -> A-layout via private LDS slice (same-wave, no barrier)
    stage4(buf, qt, lrow, lhalf);
    v16bf qa0 = make_afrag_f32(buf + lrow * 64 + 0 * 32 + kb);
    v16bf qa1 = make_afrag_f32(buf + lrow * 64 + 1 * 32 + kb);

    // ---- scores(16x64) = q_h(16x64) @ (k_h*SCALE)(64x64 padded) ----
    v8f st4[4];
    {
      const bf16* kp = kT + h * 4096 + (size_t)lrow * 64 + lhalf * 16;
#pragma unroll
      for (int st = 0; st < 4; ++st) {
        v16bf b0 = *(const v16bf*)(kp + (size_t)(st * 16) * 64);
        v16bf b1 = *(const v16bf*)(kp + (size_t)(st * 16) * 64 + 32);
        v8f c = wmma_bf16(qa0, b0, zero8());
        st4[st] = wmma_bf16(qa1, b1, c);
      }
    }

    // ---- softmax over s=0..49 (C layout: col = 16*st + lrow, row = r+8*lhalf)
    const bool v3 = (lrow < 2);  // tile 3 covers s=48..63; valid only s<50
#pragma unroll
    for (int r = 0; r < 8; ++r) {
      float s3 = v3 ? st4[3][r] : -3.0e38f;
      float mx = fmaxf(fmaxf(st4[0][r], st4[1][r]), fmaxf(st4[2][r], s3));
#pragma unroll
      for (int i = 8; i >= 1; i >>= 1) mx = fmaxf(mx, __shfl_xor(mx, i, 32));
      float e0 = __expf(st4[0][r] - mx);
      float e1 = __expf(st4[1][r] - mx);
      float e2 = __expf(st4[2][r] - mx);
      float e3 = v3 ? __expf(st4[3][r] - mx) : 0.f;
      float sm = e0 + e1 + e2 + e3;
#pragma unroll
      for (int i = 8; i >= 1; i >>= 1) sm += __shfl_xor(sm, i, 32);
      float inv = __builtin_amdgcn_rcpf(sm);
      st4[0][r] = e0 * inv;
      st4[1][r] = e1 * inv;
      st4[2][r] = e2 * inv;
      st4[3][r] = e3 * inv;
    }

    // ---- out_h(16x64) = attn(16x64) @ v_h(64x64 padded) ----
    stage4(buf, st4, lrow, lhalf);
    v16bf aa0 = make_afrag_f32(buf + lrow * 64 + 0 * 32 + kb);
    v16bf aa1 = make_afrag_f32(buf + lrow * 64 + 1 * 32 + kb);

    v8f ot[4];
    {
      const bf16* vp =
          vbuf + ((size_t)((h * 16 + batch) * 64 + lrow)) * 64 + lhalf * 16;
#pragma unroll
      for (int nt = 0; nt < 4; ++nt) {
        v16bf b0 = *(const v16bf*)(vp + (size_t)(nt * 16) * 64);
        v16bf b1 = *(const v16bf*)(vp + (size_t)(nt * 16) * 64 + 32);
        v8f c = wmma_bf16(aa0, b0, zero8());
        ot[nt] = wmma_bf16(aa1, b1, c);
      }
    }

    // ---- capture ctx_h as A-fragments (constant indices: h unrolled) ----
    stage4(buf, ot, lrow, lhalf);
    ctxa[2 * h + 0] = make_afrag_f32(buf + lrow * 64 + 0 * 32 + kb);
    ctxa[2 * h + 1] = make_afrag_f32(buf + lrow * 64 + 1 * 32 + kb);
  }

  // ---- y(16x512) = ctx(16x512) @ Wo + bo, * mask ----
  // 8 groups of 4 N-tiles; Wo group slice staged in LDS, double buffered.
#pragma unroll
  for (int g = 0; g < 8; ++g) {
    bf16* cur = (g & 1) ? wb1 : wb0;  // stage index 8+g keeps the same parity
    bf16* nxt = (g & 1) ? wb0 : wb1;
    wait_async_done();
    __syncthreads();
    if (g < 7) stage_slice(WoT + (size_t)(g + 1) * 32768, nxt, tid);

    v8f c0 = zero8(), c1 = zero8(), c2 = zero8(), c3 = zero8();
    const bf16* wbp = cur + (size_t)lrow * 512 + lhalf * 16;
#pragma unroll
    for (int kc = 0; kc < 16; ++kc) {
      v16bf b0 = *(const v16bf*)(wbp + (0 * 16) * 512 + kc * 32);
      v16bf b1 = *(const v16bf*)(wbp + (1 * 16) * 512 + kc * 32);
      v16bf b2 = *(const v16bf*)(wbp + (2 * 16) * 512 + kc * 32);
      v16bf b3 = *(const v16bf*)(wbp + (3 * 16) * 512 + kc * 32);
      c0 = wmma_bf16(ctxa[kc], b0, c0);
      c1 = wmma_bf16(ctxa[kc], b1, c1);
      c2 = wmma_bf16(ctxa[kc], b2, c2);
      c3 = wmma_bf16(ctxa[kc], b3, c3);
    }
    v8f cc[4] = {c0, c1, c2, c3};
#pragma unroll
    for (int i = 0; i < 4; ++i) {
      int nt = g * 4 + i;
      float bov = bo[nt * 16 + lrow];
#pragma unroll
      for (int r = 0; r < 8; ++r)
        out[(size_t)(rowbase + r + 8 * lhalf) * 512 + nt * 16 + lrow] =
            (cc[i][r] + bov) * mrow[r];
    }
  }
}

// ---------------- launcher ----------------

extern "C" void kernel_launch(void* const* d_in, const int* in_sizes, int n_in,
                              void* d_out, int out_size, void* d_ws, size_t ws_size,
                              hipStream_t stream) {
  (void)in_sizes; (void)n_in; (void)out_size; (void)ws_size;
  const float* x      = (const float*)d_in[0];
  const float* values = (const float*)d_in[1];
  const float* mask   = (const float*)d_in[2];
  const float* Wq     = (const float*)d_in[3];
  const float* bq     = (const float*)d_in[4];
  const float* Wv     = (const float*)d_in[5];
  const float* bv     = (const float*)d_in[6];
  const float* Wo     = (const float*)d_in[7];
  const float* bo     = (const float*)d_in[8];
  const float* keyc   = (const float*)d_in[9];

  char* ws = (char*)d_ws;
  bf16* WqT = (bf16*)(ws);                                      // 512KB
  bf16* WoT = (bf16*)(ws + 512 * 1024);                         // 512KB
  bf16* kT  = (bf16*)(ws + 1024 * 1024);                        // 64KB
  bf16* vb  = (bf16*)(ws + 1024 * 1024 + 64 * 1024);            // 1MB

  prep_weights<<<1024, 256, 0, stream>>>(Wq, Wo, WqT, WoT);
  prep_keys<<<128, 256, 0, stream>>>(keyc, kT);
  prep_values<<<2048, 256, 0, stream>>>(values, Wv, bv, vb);
  style_attn_fused<<<512, 256, 131072, stream>>>(x, mask, bq, bo, WqT, WoT, kT, vb,
                                                 (float*)d_out);
}
